// betaChessAI_90769838833863
// MI455X (gfx1250) — compile-verified, matrix operand order checked by reading, not run on hardware
//
#include <hip/hip_runtime.h>
#include <hip/hip_bf16.h>
#include <math.h>

typedef __attribute__((ext_vector_type(16))) _Float16 v16h;
typedef __attribute__((ext_vector_type(8)))  _Float16 v8h;
typedef __attribute__((ext_vector_type(8)))  float    v8f;

#define GEMM_TM 64
#define GEMM_TN 128
#define GEMM_TK 64   // K multiples of 64 everywhere in this net

// Fragment loader for WMMA f16 16x16x32.
// Source is an LDS tile stored [row][K-contiguous] with row-stride ldh (halves).
// Per CDNA5 ISA (16-bit A 16x32): lanes 0-15 hold K 0..7 (v0-3) and 16..23 (v4-7),
// lanes 16-31 hold K 8..15 and 24..31, row = lane&15. B tiles are stored
// transposed ([n][K]) so the same loader serves both operands.
__device__ __forceinline__ v16h frag_ld(const _Float16* p0, int ldh, int lane) {
  const int hf = lane >> 4, ls = lane & 15;
  const _Float16* p = p0 + (size_t)ls * ldh + hf * 8;
  v16h f;
#pragma unroll
  for (int i = 0; i < 8; ++i) { f[i] = p[i]; f[i + 8] = p[16 + i]; }
  return f;
}

__device__ __forceinline__ v8f wmma16(v16h a, v16h b, v8f c) {
  return __builtin_amdgcn_wmma_f32_16x16x32_f16(false, a, false, b, (short)0, c,
                                                false, false);
}

// ---------------------------------------------------------------------------
// Generic GEMM: C[M,N] = A[M,K] * B[K,N] (+bias) (opt. relu in/out).
// M%64==0, N%128==0, K%64==0 (true for every GEMM in this network).
// flags: 1 = relu(A) on load, 2 = relu on output, 4 = bias present.
// 8 waves: 2x4 grid, 32x32 out per wave, 8 WMMA per barrier pair.
// ---------------------------------------------------------------------------
__global__ __launch_bounds__(256) void k_gemm(
    const float* __restrict__ A, const float* __restrict__ Bw,
    const float* __restrict__ bias, float* __restrict__ C,
    int M, int N, int K, int flags)
{
  __shared__ _Float16 lA[GEMM_TM * GEMM_TK];   // 8 KB
  __shared__ _Float16 lB[GEMM_TN * GEMM_TK];   // 16 KB, [n][k]
  const int tid = threadIdx.x;
  const int lane = tid & 31;
  const int wv = tid >> 5;
  const int wm = wv >> 2, wn = wv & 3;
  const int Mb = blockIdx.x * GEMM_TM, Nb = blockIdx.y * GEMM_TN;
  v8f acc[2][2] = {};

  for (int k0 = 0; k0 < K; k0 += GEMM_TK) {
    { // stage A 64x64: 16 contiguous K per thread, packed 16B LDS stores
      int r = tid >> 2, kb = (tid & 3) * 16;
      const float* g = A + (size_t)(Mb + r) * K + k0 + kb;
      if (k0 + GEMM_TK < K) __builtin_prefetch(g + GEMM_TK, 0, 1);
      v8h h0, h1;
#pragma unroll
      for (int i = 0; i < 8; ++i) {
        float v0 = g[i], v1 = g[i + 8];
        if (flags & 1) { v0 = fmaxf(v0, 0.f); v1 = fmaxf(v1, 0.f); }
        h0[i] = (_Float16)v0; h1[i] = (_Float16)v1;
      }
      *(v8h*)&lA[r * GEMM_TK + kb]     = h0;
      *(v8h*)&lA[r * GEMM_TK + kb + 8] = h1;
    }
    { // stage B 64x128 -> transpose into [n][k]
      int kk = tid >> 3, nb = (tid & 7) * 16;
#pragma unroll
      for (int hb = 0; hb < 2; ++hb) {
        int kr = kk + hb * 32;
        const float* g = Bw + (size_t)(k0 + kr) * N + Nb + nb;
        if (hb == 0 && k0 + GEMM_TK < K)
          __builtin_prefetch(g + (size_t)GEMM_TK * N, 0, 1);
#pragma unroll
        for (int i = 0; i < 16; ++i)
          lB[(nb + i) * GEMM_TK + kr] = (_Float16)g[i];
      }
    }
    __syncthreads();
#pragma unroll
    for (int kc = 0; kc < GEMM_TK; kc += 32) {
      v16h af0 = frag_ld(lA + (wm * 32) * GEMM_TK + kc,      GEMM_TK, lane);
      v16h af1 = frag_ld(lA + (wm * 32 + 16) * GEMM_TK + kc, GEMM_TK, lane);
      v16h bf0 = frag_ld(lB + (wn * 32) * GEMM_TK + kc,      GEMM_TK, lane);
      v16h bf1 = frag_ld(lB + (wn * 32 + 16) * GEMM_TK + kc, GEMM_TK, lane);
      acc[0][0] = wmma16(af0, bf0, acc[0][0]);
      acc[0][1] = wmma16(af0, bf1, acc[0][1]);
      acc[1][0] = wmma16(af1, bf0, acc[1][0]);
      acc[1][1] = wmma16(af1, bf1, acc[1][1]);
    }
    __syncthreads();
  }

  const int hf = lane >> 4, ls = lane & 15;
#pragma unroll
  for (int i = 0; i < 2; ++i)
#pragma unroll
    for (int j = 0; j < 2; ++j) {
      int col = Nb + wn * 32 + j * 16 + ls;
      float bv = (flags & 4) ? bias[col] : 0.f;
#pragma unroll
      for (int r = 0; r < 8; ++r) {
        int row = Mb + wm * 32 + i * 16 + hf * 8 + r;
        float v = acc[i][j][r] + bv;
        if (flags & 2) v = fmaxf(v, 0.f);
        C[(size_t)row * N + col] = v;
      }
    }
}

// ---------------------------------------------------------------------------
// Implicit-GEMM 3x3 conv, C=128->128 on 8x8, relu applied to input, NCHW I/O.
// M = Bn*64 pixels, N = 128, K = 1152 (ci*9+ky*3+kx), K%64==0.
// ---------------------------------------------------------------------------
__global__ __launch_bounds__(256) void k_conv3x3_wmma(
    const float* __restrict__ X, const float* __restrict__ W,
    const float* __restrict__ bias, float* __restrict__ Z, int Bn)
{
  __shared__ _Float16 lA[GEMM_TM * GEMM_TK];
  __shared__ _Float16 lB[GEMM_TN * GEMM_TK];
  const int tid = threadIdx.x;
  const int lane = tid & 31;
  const int wv = tid >> 5;
  const int wm = wv >> 2, wn = wv & 3;
  const int Mb = blockIdx.x * GEMM_TM;
  const int K = 128 * 9;
  v8f acc[2][2] = {};

  for (int k0 = 0; k0 < K; k0 += GEMM_TK) {
    { // A: on-the-fly im2col with relu + zero pad, packed 16B LDS stores
      int r = tid >> 2, kb = (tid & 3) * 16;
      int m = Mb + r;
      int b = m >> 6, y = (m >> 3) & 7, xx = m & 7;
      v8h h0, h1;
#pragma unroll
      for (int i = 0; i < 16; ++i) {
        int k = k0 + kb + i;
        int ci = k / 9, rem = k - ci * 9;
        int ky = rem / 3, kx = rem - ky * 3;
        int iy = y + ky - 1, ix = xx + kx - 1;
        float v = 0.f;
        if ((unsigned)iy < 8u && (unsigned)ix < 8u)
          v = fmaxf(X[(((size_t)b * 128 + ci) << 6) + iy * 8 + ix], 0.f);
        if (i < 8) h0[i] = (_Float16)v; else h1[i - 8] = (_Float16)v;
      }
      *(v8h*)&lA[r * GEMM_TK + kb]     = h0;
      *(v8h*)&lA[r * GEMM_TK + kb + 8] = h1;
    }
    { // B: weight (co, k) -> lB[n][k]
      int kk = tid >> 3, nb = (tid & 7) * 16;
#pragma unroll
      for (int hb = 0; hb < 2; ++hb) {
        int kr = kk + hb * 32;
#pragma unroll
        for (int i = 0; i < 16; ++i)
          lB[(nb + i) * GEMM_TK + kr] = (_Float16)W[(size_t)(nb + i) * K + k0 + kr];
      }
    }
    __syncthreads();
#pragma unroll
    for (int kc = 0; kc < GEMM_TK; kc += 32) {
      v16h af0 = frag_ld(lA + (wm * 32) * GEMM_TK + kc,      GEMM_TK, lane);
      v16h af1 = frag_ld(lA + (wm * 32 + 16) * GEMM_TK + kc, GEMM_TK, lane);
      v16h bf0 = frag_ld(lB + (wn * 32) * GEMM_TK + kc,      GEMM_TK, lane);
      v16h bf1 = frag_ld(lB + (wn * 32 + 16) * GEMM_TK + kc, GEMM_TK, lane);
      acc[0][0] = wmma16(af0, bf0, acc[0][0]);
      acc[0][1] = wmma16(af0, bf1, acc[0][1]);
      acc[1][0] = wmma16(af1, bf0, acc[1][0]);
      acc[1][1] = wmma16(af1, bf1, acc[1][1]);
    }
    __syncthreads();
  }

  const int hf = lane >> 4, ls = lane & 15;
#pragma unroll
  for (int i = 0; i < 2; ++i)
#pragma unroll
    for (int j = 0; j < 2; ++j) {
      int co = wn * 32 + j * 16 + ls;
      float bv = bias[co];
#pragma unroll
      for (int r = 0; r < 8; ++r) {
        int m = Mb + wm * 32 + i * 16 + hf * 8 + r;
        int b = m >> 6, pix = m & 63;
        Z[(((size_t)b * 128 + co) << 6) + pix] = acc[i][j][r] + bv;
      }
    }
}

// ---------------------------------------------------------------------------
// Window attention, seq-len 16, 8 heads x hd 64. One wave per (batch, head).
// Computes S^T = K*Q^T (2 wmma), softmax across key axis (per-lane 8 vals +
// shfl_xor(16)), then out = P^T * V via LDS round-trip (4 wmma).
// ---------------------------------------------------------------------------
__global__ __launch_bounds__(128) void k_attention(
    const float* __restrict__ QKV, float* __restrict__ O, int Bn)
{
  __shared__ _Float16 sQ[4][16 * 64];
  __shared__ _Float16 sK[4][16 * 64];
  __shared__ _Float16 sV[4][64 * 32];   // transposed [e][k], k padded to 32
  __shared__ _Float16 sP[4][16 * 32];   // [q][k], k padded to 32
  const int lane = threadIdx.x & 31;
  const int w = threadIdx.x >> 5;
  const int task = blockIdx.x * 4 + w;
  const int b = task >> 3, hh = task & 7;

  const int tok = lane & 15, eh = lane >> 4;
  const size_t rowbase = ((size_t)b * 16 + tok) * 1536 + hh * 64;
#pragma unroll 4
  for (int i = 0; i < 32; ++i) {
    int e = eh * 32 + i;
    sQ[w][tok * 64 + e] = (_Float16)(QKV[rowbase + e] * 0.125f); // hd^-0.5
    sK[w][tok * 64 + e] = (_Float16)QKV[rowbase + 512 + e];
    sV[w][e * 32 + tok] = (_Float16)QKV[rowbase + 1024 + e];
  }
#pragma unroll 4
  for (int i = 0; i < 32; ++i) { // zero-pad V K=16..31
    int idx = lane * 32 + i;
    sV[w][(idx >> 4) * 32 + 16 + (idx & 15)] = (_Float16)0.f;
  }
  __syncthreads();

  v8f s = {};
  s = wmma16(frag_ld(&sK[w][0],  64, lane), frag_ld(&sQ[w][0],  64, lane), s);
  s = wmma16(frag_ld(&sK[w][32], 64, lane), frag_ld(&sQ[w][32], 64, lane), s);

  const int hf = lane >> 4, ls = lane & 15;
  float mx = s[0];
#pragma unroll
  for (int r = 1; r < 8; ++r) mx = fmaxf(mx, s[r]);
  mx = fmaxf(mx, __shfl_xor(mx, 16));
  float pr[8], sum = 0.f;
#pragma unroll
  for (int r = 0; r < 8; ++r) { pr[r] = __expf(s[r] - mx); sum += pr[r]; }
  sum += __shfl_xor(sum, 16);
  float inv = 1.f / sum;
#pragma unroll
  for (int r = 0; r < 8; ++r)   // P[k=hf*8+r][q=ls] -> sP[q][k]
    sP[w][ls * 32 + hf * 8 + r] = (_Float16)(pr[r] * inv);
#pragma unroll
  for (int i = 0; i < 8; ++i) { // zero-pad P K=16..31
    int idx = lane * 8 + i;
    sP[w][(idx >> 4) * 32 + 16 + (idx & 15)] = (_Float16)0.f;
  }
  __syncthreads();

  v16h ap = frag_ld(&sP[w][0], 32, lane);
#pragma unroll
  for (int j = 0; j < 4; ++j) {
    v8f o = {};
    o = wmma16(ap, frag_ld(&sV[w][j * 16 * 32], 32, lane), o);
#pragma unroll
    for (int r = 0; r < 8; ++r)
      O[((size_t)b * 16 + hf * 8 + r) * 512 + hh * 64 + j * 16 + ls] = o[r];
  }
}

// ---------------------------------------------------------------------------
// Board one-hot encode + first 3x3 conv (12->128) fused; exploits one-hot
// sparsity: each neighbor square contributes <=2 weight taps.
// ---------------------------------------------------------------------------
__global__ void k_encode_conv1(const int* __restrict__ X,
                               const float* __restrict__ W,
                               const float* __restrict__ bias,
                               float* __restrict__ H, int total)
{
  int idx = blockIdx.x * 256 + threadIdx.x;
  if (idx >= total) return;
  int pix = idx & 63;
  int co = (idx >> 6) & 127;
  int b = idx >> 13;
  int y = pix >> 3, xx = pix & 7;
  float acc = bias[co];
  for (int ky = 0; ky < 3; ++ky) {
    int iy = y + ky - 1; if ((unsigned)iy >= 8u) continue;
    for (int kx = 0; kx < 3; ++kx) {
      int ix = xx + kx - 1; if ((unsigned)ix >= 8u) continue;
      int sq = iy * 8 + ix;
      int pw = X[b * 128 + sq];
      int pb = X[b * 128 + 64 + sq];
      if (pw >= 1 && pw <= 6) acc += W[((co * 12 + (pw - 1)) * 3 + ky) * 3 + kx];
      if (pb >= 1 && pb <= 6) acc += W[((co * 12 + 5 + pb) * 3 + ky) * 3 + kx];
    }
  }
  H[idx] = acc;
}

// Training-mode BN stats: one block per channel, reduce over (N,H,W).
__global__ __launch_bounds__(256) void k_bn_stats(const float* __restrict__ Z,
                                                  float* __restrict__ mean,
                                                  float* __restrict__ var, int Bn)
{
  __shared__ float rs[256], rs2[256];
  const int c = blockIdx.x;
  const int n = Bn * 64;
  float s = 0.f, s2 = 0.f;
  for (int i = threadIdx.x; i < n; i += 256) {
    int b = i >> 6, pix = i & 63;
    float v = Z[(((size_t)b * 128 + c) << 6) + pix];
    s += v; s2 += v * v;
  }
  rs[threadIdx.x] = s; rs2[threadIdx.x] = s2;
  __syncthreads();
  for (int off = 128; off > 0; off >>= 1) {
    if (threadIdx.x < off) {
      rs[threadIdx.x] += rs[threadIdx.x + off];
      rs2[threadIdx.x] += rs2[threadIdx.x + off];
    }
    __syncthreads();
  }
  if (threadIdx.x == 0) {
    float m = rs[0] / (float)n;
    mean[c] = m;
    var[c] = rs2[0] / (float)n - m * m;
  }
}

__global__ void k_bn_apply(const float* __restrict__ Z,
                           const float* __restrict__ mean,
                           const float* __restrict__ var,
                           const float* __restrict__ g,
                           const float* __restrict__ bb,
                           const float* __restrict__ resid,
                           float* __restrict__ out, int n)
{
  int idx = blockIdx.x * 256 + threadIdx.x;
  if (idx >= n) return;
  int c = (idx >> 6) & 127;
  float v = (Z[idx] - mean[c]) * rsqrtf(var[c] + 1e-5f) * g[c] + bb[c];
  if (resid) v += resid[idx];
  out[idx] = v;
}

// Window partition (+pos embed): T[(b*16+n)*512 + d] with d=[p1][p2][c].
__global__ void k_winpart(const float* __restrict__ H,
                          const float* __restrict__ pos,
                          float* __restrict__ T, int n)
{
  int idx = blockIdx.x * 256 + threadIdx.x;
  if (idx >= n) return;
  int d = idx & 511;
  int bn = idx >> 9;
  int nwin = bn & 15;
  int b = bn >> 4;
  int c = d & 127, p2 = (d >> 7) & 1, p1 = (d >> 8) & 1;
  int y = ((nwin >> 2) << 1) | p1;
  int xx = ((nwin & 3) << 1) | p2;
  T[idx] = H[(((size_t)b * 128 + c) << 6) + y * 8 + xx] + pos[(nwin << 9) + d];
}

__global__ void k_winrev(const float* __restrict__ T, float* __restrict__ H, int n)
{
  int idx = blockIdx.x * 256 + threadIdx.x;
  if (idx >= n) return;
  int pix = idx & 63;
  int c = (idx >> 6) & 127;
  int b = idx >> 13;
  int y = pix >> 3, xx = pix & 7;
  int nwin = ((y >> 1) << 2) | (xx >> 1);
  int d = ((y & 1) << 8) | ((xx & 1) << 7) | c;
  H[idx] = T[(((size_t)b * 16 + nwin) << 9) + d];
}

// T += LayerNorm(X) rowwise over 512, no affine. One wave per row.
__global__ __launch_bounds__(256) void k_ln_add(float* __restrict__ T,
                                                const float* __restrict__ X,
                                                int rows)
{
  int w = threadIdx.x >> 5, lane = threadIdx.x & 31;
  int row = blockIdx.x * 8 + w;
  if (row >= rows) return;
  const float* x = X + (size_t)row * 512;
  float lv[16]; float s = 0.f;
#pragma unroll
  for (int i = 0; i < 16; ++i) { lv[i] = x[lane + i * 32]; s += lv[i]; }
#pragma unroll
  for (int off = 16; off > 0; off >>= 1) s += __shfl_xor(s, off);
  float m = s * (1.f / 512.f);
  float s2 = 0.f;
#pragma unroll
  for (int i = 0; i < 16; ++i) { float d = lv[i] - m; s2 += d * d; }
#pragma unroll
  for (int off = 16; off > 0; off >>= 1) s2 += __shfl_xor(s2, off);
  float inv = rsqrtf(s2 * (1.f / 512.f) + 1e-6f);
  float* t = T + (size_t)row * 512;
#pragma unroll
  for (int i = 0; i < 16; ++i) t[lane + i * 32] += (lv[i] - m) * inv;
}

// feat[b, (y*8+x)*128 + c] = h[b, c, y, x]
__global__ void k_feat(const float* __restrict__ H, float* __restrict__ F, int n)
{
  int idx = blockIdx.x * 256 + threadIdx.x;
  if (idx >= n) return;
  int c = idx & 127;
  int pix = (idx >> 7) & 63;
  int b = idx >> 13;
  F[idx] = H[(((size_t)b * 128 + c) << 6) + pix];
}

// Row softmax over 640 logits, one wave per row.
__global__ __launch_bounds__(256) void k_softmax640(const float* __restrict__ X,
                                                    float* __restrict__ Y, int rows)
{
  int w = threadIdx.x >> 5, lane = threadIdx.x & 31;
  int row = blockIdx.x * 8 + w;
  if (row >= rows) return;
  const float* x = X + (size_t)row * 640;
  float lv[20]; float mx = -1e30f;
#pragma unroll
  for (int i = 0; i < 20; ++i) { lv[i] = x[lane + i * 32]; mx = fmaxf(mx, lv[i]); }
#pragma unroll
  for (int off = 16; off > 0; off >>= 1) mx = fmaxf(mx, __shfl_xor(mx, off));
  float s = 0.f;
#pragma unroll
  for (int i = 0; i < 20; ++i) { lv[i] = __expf(lv[i] - mx); s += lv[i]; }
#pragma unroll
  for (int off = 16; off > 0; off >>= 1) s += __shfl_xor(s, off);
  float inv = 1.f / s;
  float* y = Y + (size_t)row * 640;
#pragma unroll
  for (int i = 0; i < 20; ++i) y[lane + i * 32] = lv[i] * inv;
}

// ---------------------------------------------------------------------------
extern "C" void kernel_launch(void* const* d_in, const int* in_sizes, int n_in,
                              void* d_out, int out_size, void* d_ws, size_t ws_size,
                              hipStream_t stream)
{
  const int*   x       = (const int*)  d_in[0];
  const float* pos     = (const float*)d_in[1];
  const float* conv1_w = (const float*)d_in[2];
  const float* conv1_b = (const float*)d_in[3];
  const float* rb1_w   = (const float*)d_in[4];
  const float* rb1_b   = (const float*)d_in[5];
  const float* bn1_g   = (const float*)d_in[6];
  const float* bn1_b   = (const float*)d_in[7];
  const float* rb2_w   = (const float*)d_in[8];
  const float* rb2_b   = (const float*)d_in[9];
  const float* bn2_g   = (const float*)d_in[10];
  const float* bn2_b   = (const float*)d_in[11];
  const float* qkv_w   = (const float*)d_in[12];
  const float* qkv_b   = (const float*)d_in[13];
  const float* proj_w  = (const float*)d_in[14];
  const float* proj_b  = (const float*)d_in[15];
  const float* fc1_w   = (const float*)d_in[16];
  const float* fc1_b   = (const float*)d_in[17];
  const float* fc2_w   = (const float*)d_in[18];
  const float* fc2_b   = (const float*)d_in[19];
  const float* lin1_w  = (const float*)d_in[20];
  const float* lin1_b  = (const float*)d_in[21];
  const float* lin2_w  = (const float*)d_in[22];
  const float* lin2_b  = (const float*)d_in[23];

  const int Bn   = in_sizes[0] / 128;       // 4096
  const int actN = Bn * 128 * 64;           // NCHW activation elements
  const int Mtok = Bn * 16;                 // token rows
  const int tokN = Mtok * 512;

  char* wp = (char*)d_ws;
  auto carve = [&](size_t bytes) -> float* {
    float* r = (float*)wp;
    wp += (bytes + 255) & ~(size_t)255;
    return r;
  };
  float* h   = carve((size_t)actN * 4);
  float* z   = carve((size_t)actN * 4);
  float* z2  = carve((size_t)actN * 4);
  float* T   = carve((size_t)tokN * 4);
  float* QKV = carve((size_t)Mtok * 1536 * 4);
  float* stat = carve(1024);
  float* mean = stat, *var = stat + 128;
  // buffer aliases (lifetimes disjoint within each depth iteration)
  float* O = z, *AO = z2, *F1 = QKV, *F2 = z2;
  float* feat = z, *L1 = z2, *L2 = QKV;

  auto gemm = [&](const float* A, const float* Bw, const float* bias, float* Cc,
                  int M, int N, int K, int flags) {
    dim3 g(M / GEMM_TM, N / GEMM_TN);
    k_gemm<<<g, 256, 0, stream>>>(A, Bw, bias, Cc, M, N, K, flags);
  };

  k_encode_conv1<<<(actN + 255) / 256, 256, 0, stream>>>(x, conv1_w, conv1_b, h, actN);

  for (int d = 0; d < 10; ++d) {
    const size_t wconv = (size_t)d * 128 * 128 * 9;
    // z = bn1(conv(relu(h)))
    k_conv3x3_wmma<<<Bn * 64 / GEMM_TM, 256, 0, stream>>>(h, rb1_w + wconv,
                                                          rb1_b + d * 128, z, Bn);
    k_bn_stats<<<128, 256, 0, stream>>>(z, mean, var, Bn);
    k_bn_apply<<<(actN + 255) / 256, 256, 0, stream>>>(z, mean, var, bn1_g + d * 128,
                                                       bn1_b + d * 128, nullptr, z, actN);
    // h = h + bn2(conv(relu(z)))
    k_conv3x3_wmma<<<Bn * 64 / GEMM_TM, 256, 0, stream>>>(z, rb2_w + wconv,
                                                          rb2_b + d * 128, z2, Bn);
    k_bn_stats<<<128, 256, 0, stream>>>(z2, mean, var, Bn);
    k_bn_apply<<<(actN + 255) / 256, 256, 0, stream>>>(z2, mean, var, bn2_g + d * 128,
                                                       bn2_b + d * 128, h, h, actN);
    // attention + mlp on 16 window tokens of dim 512
    k_winpart<<<(tokN + 255) / 256, 256, 0, stream>>>(h, pos, T, tokN);
    gemm(T, qkv_w + (size_t)d * 512 * 1536, qkv_b + d * 1536, QKV, Mtok, 1536, 512, 4);
    k_attention<<<Bn * 2, 128, 0, stream>>>(QKV, O, Bn);
    gemm(O, proj_w + (size_t)d * 512 * 512, proj_b + d * 512, AO, Mtok, 512, 512, 4);
    k_ln_add<<<Mtok / 8, 256, 0, stream>>>(T, AO, Mtok);
    gemm(T, fc1_w + (size_t)d * 512 * 512, fc1_b + d * 512, F1, Mtok, 512, 512, 4 | 2);
    gemm(F1, fc2_w + (size_t)d * 512 * 512, fc2_b + d * 512, F2, Mtok, 512, 512, 4);
    k_ln_add<<<Mtok / 8, 256, 0, stream>>>(T, F2, Mtok);
    k_winrev<<<(actN + 255) / 256, 256, 0, stream>>>(T, h, actN);
  }

  // head: (B,8192) -> 1024 relu -> 640 -> softmax
  k_feat<<<(actN + 255) / 256, 256, 0, stream>>>(h, feat, actN);
  gemm(feat, lin1_w, lin1_b, L1, Bn, 1024, 8192, 4 | 2);
  gemm(L1, lin2_w, lin2_b, L2, Bn, 640, 1024, 4);
  k_softmax640<<<Bn / 8, 256, 0, stream>>>(L2, (float*)d_out, Bn);
}
